// LSTMPredictor_3221225472591
// MI455X (gfx1250) — matrix operand order repaired and checked by reading, not compiled, not run on hardware
//
#include <hip/hip_runtime.h>
#include <math.h>

typedef __bf16 bf16_t;
typedef __attribute__((ext_vector_type(16))) __bf16 v16bf;
typedef __attribute__((ext_vector_type(8)))  float  v8f;

constexpr int kB = 256, kT = 512, kI = 64, kH = 512;
constexpr int K0  = kI + kH;     // 576  (x ; h0)
constexpr int K1  = kH + kH;     // 1024 (h1 ; h2)
constexpr int KP0 = K0 + 8;      // 584  row pad (8 elems) -> uniform bank spread
constexpr int KP1 = K1 + 8;      // 1032
constexpr int NWG = 64;
constexpr int NTH = 256;

// workspace layout (bytes)
constexpr size_t SZ_H    = (size_t)kB * kH * sizeof(bf16_t);      // 256 KB
constexpr size_t OFF_CNT = 0;                                     // 2048 ints
constexpr size_t OFF_H0  = 8192;                                  // double buffered
constexpr size_t OFF_H1  = OFF_H0 + 2 * SZ_H;                     // double buffered
constexpr size_t OFF_H2  = OFF_H1 + 2 * SZ_H;                     // double buffered
constexpr size_t OFF_XBF = OFF_H2 + 2 * SZ_H;                     // [T][B][I] bf16, 16 MB
// total ~17.5 MB

union Frag { v16bf v; uint4 q[2]; };

__device__ inline float sigm(float x) { return 1.0f / (1.0f + __expf(-x)); }

// A fragment (16x32 bf16): lanes 0-15 hold K {0..7, 16..23}, lanes 16-31 {8..15, 24..31}
__device__ inline void loadA(Frag& f, const bf16_t* p) {
  f.q[0] = *(const uint4*)(p);        // K = ks .. ks+7
  f.q[1] = *(const uint4*)(p + 16);   // K = 16+ks .. 16+ks+7
}

// B fragments (32x16 bf16): lanes 0-15 hold K {0..7, 16..23}, lanes 16-31 {8..15, 24..31}.
// 4 gate WMMAs share one A fragment; B comes from LDS-resident weights (padded rows).
__device__ inline void wmma4(v8f acc[4], const Frag& a, const bf16_t* lw,
                             int n, int k0, int kb, int kp) {
#pragma unroll
  for (int q = 0; q < 4; ++q) {
    Frag bfr;
    const bf16_t* p = lw + ((size_t)(q * 16 + n) * kp + k0 + kb);
    bfr.q[0] = *(const uint4*)(p);      // K = kb .. kb+7
    bfr.q[1] = *(const uint4*)(p + 16); // K = 16+kb .. 16+kb+7
    acc[q] = __builtin_amdgcn_wmma_f32_16x16x32_bf16(
        false, a.v, false, bfr.v, (short)0, acc[q], false, false);
  }
}

__device__ inline void grid_sync(int* cnt, int& s) {
  __builtin_amdgcn_fence(__ATOMIC_RELEASE, "agent");
  __syncthreads();
  if (threadIdx.x == 0) {
    __hip_atomic_fetch_add(&cnt[s], 1, __ATOMIC_RELEASE, __HIP_MEMORY_SCOPE_AGENT);
    while (__hip_atomic_load(&cnt[s], __ATOMIC_ACQUIRE, __HIP_MEMORY_SCOPE_AGENT) < NWG) {
      __builtin_amdgcn_s_sleep(1);
    }
  }
  __syncthreads();
  __builtin_amdgcn_fence(__ATOMIC_ACQUIRE, "agent");
  ++s;
}

__global__ void lstm_init(char* __restrict__ ws) {
  int gt = blockIdx.x * blockDim.x + threadIdx.x;
  int stride = gridDim.x * blockDim.x;
  int* cnt = (int*)(ws + OFF_CNT);
  for (int i = gt; i < 2048; i += stride) cnt[i] = 0;
  uint32_t* h0 = (uint32_t*)(ws + OFF_H0);          // zero ping buffer 0
  uint32_t* h2 = (uint32_t*)(ws + OFF_H2);
  for (int i = gt; i < kB * kH / 2; i += stride) { h0[i] = 0u; h2[i] = 0u; }
}

__global__ __launch_bounds__(NTH, 1)
void lstm_fused(const float* __restrict__ x,
                const float* __restrict__ Wih0, const float* __restrict__ Whh0,
                const float* __restrict__ bih0, const float* __restrict__ bhh0,
                const float* __restrict__ Wih1, const float* __restrict__ Whh1,
                const float* __restrict__ bih1, const float* __restrict__ bhh1,
                const float* __restrict__ Wfc,  const float* __restrict__ bfc,
                float* __restrict__ out, char* __restrict__ ws) {
  // resident weights: [gate q][j 0..16)][k 0..K) bf16, padded rows
  __shared__ __align__(16) bf16_t lw0[64 * KP0];   //  73 KB
  __shared__ __align__(16) bf16_t lw1[64 * KP1];   // 129 KB

  int*    cnt = (int*)(ws + OFF_CNT);
  bf16_t* h0b = (bf16_t*)(ws + OFF_H0);
  bf16_t* h1b = (bf16_t*)(ws + OFF_H1);
  bf16_t* h2b = (bf16_t*)(ws + OFF_H2);
  bf16_t* xbf = (bf16_t*)(ws + OFF_XBF);

  const int tid   = threadIdx.x;
  const int wg    = blockIdx.x;        // 64 wgs: (jt 0..31) x (bg 0..1)
  const int wave  = tid >> 5;
  const int lane  = tid & 31;
  const int n     = lane & 15;         // A row / B col / C col within tile
  const int hi    = lane >> 4;
  const int ks    = hi ? 8 : 0;        // A-matrix K sub-select
  const int kb    = hi ? 8 : 0;        // B-matrix K sub-select
  const int jbase = (wg >> 1) * 16;
  const int bbase = (wg & 1) * 128 + wave * 16;

  // ---- one-time: x (B,T,I) f32 -> xbf (T,B,I) bf16 ----
  {
    int gt = wg * NTH + tid;
    for (int idx = gt; idx < kT * kB * kI; idx += NWG * NTH) {
      int i = idx & (kI - 1);
      int r = idx >> 6;
      int b = r & (kB - 1);
      int t = r >> 8;
      xbf[idx] = (bf16_t)x[((size_t)b * kT + t) * kI + i];
    }
  }

  // ---- one-time: weights -> LDS (bf16, concatenated K = [in ; hh]) ----
  for (int e = tid; e < 64 * K0; e += NTH) {
    int k = e % K0, r = e / K0;
    int g = (r >> 4) * kH + jbase + (r & 15);
    float w = (k < kI) ? Wih0[(size_t)g * kI + k] : Whh0[(size_t)g * kH + (k - kI)];
    lw0[(size_t)r * KP0 + k] = (bf16_t)w;
  }
  for (int e = tid; e < 64 * K1; e += NTH) {
    int k = e % K1, r = e / K1;
    int g = (r >> 4) * kH + jbase + (r & 15);
    float w = (k < kH) ? Wih1[(size_t)g * kH + k] : Whh1[(size_t)g * kH + (k - kH)];
    lw1[(size_t)r * KP1 + k] = (bf16_t)w;
  }
  __syncthreads();

  float bias0[4], bias1[4];
#pragma unroll
  for (int q = 0; q < 4; ++q) {
    int g = q * kH + jbase + n;
    bias0[q] = bih0[g] + bhh0[g];
    bias1[q] = bih1[g] + bhh1[g];
  }

  v8f c0 = {}, c2 = {};        // cell states live in registers for all 512 steps
  int s = 0;
  grid_sync(cnt, s);           // xbf visible everywhere

  // Software-pipelined over layers: tick u runs layer0 step u and layer1 step
  // u-1 concurrently (no dependence between them), then ONE device barrier.
  for (int u = 0; u <= kT; ++u) {
    // ============ layer 0, step t0 = u : gates = [x_t ; h0] W^T + b ==========
    if (u < kT) {
      const int rp = u & 1, wp = rp ^ 1;
      v8f acc[4];
#pragma unroll
      for (int q = 0; q < 4; ++q)
#pragma unroll
        for (int e = 0; e < 8; ++e) acc[q][e] = bias0[q];

      const bf16_t* xrow = xbf + ((size_t)u * kB + bbase + n) * kI + ks;
      const bf16_t* hrow = h0b + (size_t)rp * kB * kH + (size_t)(bbase + n) * kH + ks;

#pragma unroll
      for (int k0 = 0; k0 < kI; k0 += 32) {
        Frag a; loadA(a, xrow + k0);
        wmma4(acc, a, lw0, n, k0, kb, KP0);
      }
      for (int k0 = 0; k0 < kH; k0 += 32) {
        Frag a; loadA(a, hrow + k0);
        wmma4(acc, a, lw0, n, kI + k0, kb, KP0);
      }

      v8f hv;
#pragma unroll
      for (int e = 0; e < 8; ++e) {
        float ig = sigm(acc[0][e]);
        float fg = sigm(acc[1][e]);
        float gg = tanhf(acc[2][e]);
        float og = sigm(acc[3][e]);
        float c  = fg * c0[e] + ig * gg;
        c0[e] = c;
        hv[e] = og * tanhf(c);
      }
      bf16_t* w0 = h0b + (size_t)wp * kB * kH;
      bf16_t* w1 = h1b + (size_t)(u & 1) * kB * kH;   // h1 ping-pong on tick
#pragma unroll
      for (int e = 0; e < 8; ++e) {
        int b = bbase + hi * 8 + e;
        bf16_t v = (bf16_t)hv[e];
        w0[(size_t)b * kH + jbase + n] = v;
        w1[(size_t)b * kH + jbase + n] = v;
      }
    }

    // ============ layer 1, step t1 = u-1 : gates = [h1 ; h2] W^T + b =========
    if (u > 0) {
      const int t1 = u - 1;
      const int rp = t1 & 1, wp = rp ^ 1;
      v8f acc[4];
#pragma unroll
      for (int q = 0; q < 4; ++q)
#pragma unroll
        for (int e = 0; e < 8; ++e) acc[q][e] = bias1[q];

      const bf16_t* h1row = h1b + (size_t)(t1 & 1) * kB * kH + (size_t)(bbase + n) * kH + ks;
      const bf16_t* h2row = h2b + (size_t)rp * kB * kH + (size_t)(bbase + n) * kH + ks;

      for (int k0 = 0; k0 < kH; k0 += 32) {
        Frag a; loadA(a, h1row + k0);
        wmma4(acc, a, lw1, n, k0, kb, KP1);
      }
      for (int k0 = 0; k0 < kH; k0 += 32) {
        Frag a; loadA(a, h2row + k0);
        wmma4(acc, a, lw1, n, kH + k0, kb, KP1);
      }

      v8f hv;
#pragma unroll
      for (int e = 0; e < 8; ++e) {
        float ig = sigm(acc[0][e]);
        float fg = sigm(acc[1][e]);
        float gg = tanhf(acc[2][e]);
        float og = sigm(acc[3][e]);
        float c  = fg * c2[e] + ig * gg;
        c2[e] = c;
        hv[e] = og * tanhf(c);
      }
      bf16_t* w2 = h2b + (size_t)wp * kB * kH;
#pragma unroll
      for (int e = 0; e < 8; ++e) {
        int b = bbase + hi * 8 + e;
        w2[(size_t)b * kH + jbase + n] = (bf16_t)hv[e];
      }
    }

    grid_sync(cnt, s);   // one barrier per tick (513 total)
  }

  // ================= final FC: out[b] = h2_last . W_fc + b_fc ================
  if (wg == 0) {
    const bf16_t* hf = h2b + (size_t)(kT & 1) * kB * kH;  // T even -> buffer 0
    int b = tid;
    if (b < kB) {
      float sum = bfc[0];
      for (int j = 0; j < kH; ++j) sum += (float)hf[(size_t)b * kH + j] * Wfc[j];
      out[b] = sum;
    }
  }
}

extern "C" void kernel_launch(void* const* d_in, const int* in_sizes, int n_in,
                              void* d_out, int out_size, void* d_ws, size_t ws_size,
                              hipStream_t stream) {
  (void)in_sizes; (void)n_in; (void)out_size; (void)ws_size;
  const float* x    = (const float*)d_in[0];
  const float* Wih0 = (const float*)d_in[1];
  const float* Whh0 = (const float*)d_in[2];
  const float* bih0 = (const float*)d_in[3];
  const float* bhh0 = (const float*)d_in[4];
  const float* Wih1 = (const float*)d_in[5];
  const float* Whh1 = (const float*)d_in[6];
  const float* bih1 = (const float*)d_in[7];
  const float* bhh1 = (const float*)d_in[8];
  const float* Wfc  = (const float*)d_in[9];
  const float* bfc  = (const float*)d_in[10];
  char* ws = (char*)d_ws;

  lstm_init<<<NWG, NTH, 0, stream>>>(ws);
  lstm_fused<<<NWG, NTH, 0, stream>>>(x, Wih0, Whh0, bih0, bhh0,
                                      Wih1, Whh1, bih1, bhh1,
                                      Wfc, bfc, (float*)d_out, ws);
}
